// Tplink_56375740727597
// MI455X (gfx1250) — matrix operand order repaired and checked by reading, not compiled, not run on hardware
//
#include <hip/hip_runtime.h>
#include <hip/hip_bf16.h>

typedef __attribute__((ext_vector_type(16))) _Float16 v16h;
typedef __attribute__((ext_vector_type(8)))  float    v8f;

#define LSEQ 256
#define HID  256
#define GATE 1024   // 4*HID
#define ENCD 512    // 2*HID

// ---------------------------------------------------------------------------
// Fragment loaders (ISA 7.12.2 layouts, wave32)
// ---------------------------------------------------------------------------
__device__ inline v16h load_a_frag(const _Float16* __restrict__ A, int lda,
                                   int m0, int k0, int lane) {
    int m    = m0 + (lane & 15);
    int kofs = (lane & 16) ? 8 : 0;
    const _Float16* p = A + (size_t)m * lda + k0 + kofs;
    union { uint4 q[2]; v16h v; } u;
    u.q[0] = *(const uint4*)(p);        // K = kofs .. kofs+7
    u.q[1] = *(const uint4*)(p + 16);   // K = kofs+16 .. kofs+23
    return u.v;
}

// B pre-packed fragment-major: frag (kt,nt) = 32 lanes x 16 halves contiguous.
__device__ inline v16h load_b_packed(const _Float16* __restrict__ Bp,
                                     int tilesN, int kt, int nt, int lane) {
    const _Float16* p = Bp + (((size_t)(kt * tilesN + nt) * 32) + lane) * 16;
    union { uint4 q[2]; v16h v; } u;
    u.q[0] = *(const uint4*)(p);
    u.q[1] = *(const uint4*)(p + 8);
    return u.v;
}

__device__ inline float sigm(float x) { return 1.0f / (1.0f + __expf(-x)); }

// ---------------------------------------------------------------------------
// Pack f32 weight (K x N, zero-pad to Npad) into f16 fragment-major layout.
// ---------------------------------------------------------------------------
__global__ void k_pack_b(const float* __restrict__ src, _Float16* __restrict__ dst,
                         int K, int N, int Npad) {
    int idx = blockIdx.x * blockDim.x + threadIdx.x;
    if (idx >= K * Npad) return;
    int k = idx / Npad, n = idx % Npad;
    float v = (n < N) ? src[(size_t)k * N + n] : 0.0f;
    int tilesN = Npad >> 4;
    int kt = k >> 5, kr = k & 31;
    int nt = n >> 4;
    int lane = (((kr >> 3) & 1) << 4) + (n & 15);
    int i    = (kr & 7) + ((kr >> 4) << 3);
    dst[(((size_t)(kt * tilesN + nt) * 32) + lane) * 16 + i] = (_Float16)v;
}

// ---------------------------------------------------------------------------
// Generic f16 WMMA GEMM: C(MxN) = A(MxK) @ Bpacked(KxN) [+bias] [relu]
// K must be a multiple of 64 (two fragments batched per iteration).
// ---------------------------------------------------------------------------
template <bool RELU, bool HASBIAS, bool HASC, bool HASCH>
__global__ void k_gemm_f16(const _Float16* __restrict__ A,
                           const _Float16* __restrict__ Bp,
                           const float* __restrict__ bias,
                           float* __restrict__ C, _Float16* __restrict__ Ch,
                           int M, int N, int K) {
    int lane = threadIdx.x & 31;
    int wid  = (blockIdx.x * blockDim.x + threadIdx.x) >> 5;
    int tilesN = N >> 4;
    int tiles  = (M >> 4) * tilesN;
    if (wid >= tiles) return;
    int mt = wid / tilesN, nt = wid % tilesN;
    int m0 = mt << 4;

    v8f acc = {};
    for (int k2 = 0; k2 < (K >> 6); ++k2) {
        int kt = k2 * 2;
        v16h a0 = load_a_frag(A, K, m0, kt << 5, lane);
        v16h b0 = load_b_packed(Bp, tilesN, kt, nt, lane);
        v16h a1 = load_a_frag(A, K, m0, (kt + 1) << 5, lane);
        v16h b1 = load_b_packed(Bp, tilesN, kt + 1, nt, lane);
        acc = __builtin_amdgcn_wmma_f32_16x16x32_f16(false, a0, false, b0,
                                                     (short)0, acc, false, false);
        acc = __builtin_amdgcn_wmma_f32_16x16x32_f16(false, a1, false, b1,
                                                     (short)0, acc, false, false);
    }
    int n  = (nt << 4) + (lane & 15);
    int mb = m0 + ((lane & 16) ? 8 : 0);
    float bv = HASBIAS ? bias[n] : 0.0f;
#pragma unroll
    for (int r = 0; r < 8; ++r) {
        float v = acc[r] + bv;
        if (RELU) v = fmaxf(v, 0.0f);
        size_t o = (size_t)(mb + r) * N + n;
        if (HASC)  C[o]  = v;
        if (HASCH) Ch[o] = (_Float16)v;
    }
}

// ---------------------------------------------------------------------------
// Embedding gather: x[b,t,:] = [char_embed | word_embed], stored as f16
// ---------------------------------------------------------------------------
__global__ void k_embed(const int* __restrict__ inputs, const int* __restrict__ words,
                        const float* __restrict__ et, const float* __restrict__ wt,
                        _Float16* __restrict__ xh) {
    int idx = blockIdx.x * blockDim.x + threadIdx.x;
    if (idx >= 2 * LSEQ * 256) return;
    int row = idx >> 8;      // b*L + t
    int d   = idx & 255;
    float v = (d < 128) ? et[(size_t)inputs[row] * 128 + d]
                        : wt[(size_t)words[row]  * 128 + (d - 128)];
    xh[idx] = (_Float16)v;
}

// ---------------------------------------------------------------------------
// Persistent BiLSTM recurrence: blockIdx.x = direction (0 fwd, 1 bwd).
// 512 threads = 16 waves; each wave owns 4 n-tiles (64 gate columns).
// - U fragments double-buffered: tile nt+1 loads issue while tile nt's 8
//   WMMAs execute (global_load_b128, L2-hot; opaque *offset* defeats LICM
//   without demoting to flat).
// - xW[t] loaded into registers at step start (hidden behind WMMA chain).
// - mask preloaded into LDS once.
// ---------------------------------------------------------------------------
__global__ void __launch_bounds__(512)
k_lstm(const _Float16* __restrict__ Ufwd, const _Float16* __restrict__ Ubwd,
       const float* __restrict__ xWfwd, const float* __restrict__ xWbwd,
       const int* __restrict__ inputs, _Float16* __restrict__ enc_h) {
    const int dir = blockIdx.x;
    const _Float16* U  = dir ? Ubwd  : Ufwd;   // packed, tilesN = 64
    const float*    xw = dir ? xWbwd : xWfwd;

    __shared__ alignas(16) float    hsf[2][HID];
    __shared__ alignas(16) float    csf[2][HID];
    __shared__ alignas(16) _Float16 hh16[16][HID];   // rows 2..15 always zero
    __shared__ alignas(16) float    zb[2][GATE];
    __shared__ unsigned char        msk[2][LSEQ];

    int tid  = threadIdx.x;
    int lane = tid & 31;
    int wid  = tid >> 5;

    for (int i = tid; i < 16 * HID; i += 512) hh16[i >> 8][i & 255] = (_Float16)0.0f;
    { int b = tid >> 8, k = tid & 255;
      hsf[b][k] = 0.0f; csf[b][k] = 0.0f;
      msk[b][k] = (inputs[b * LSEQ + k] != 0) ? 1 : 0; }
    __syncthreads();

    for (int s = 0; s < LSEQ; ++s) {
        int t = dir ? (LSEQ - 1 - s) : s;

        // Opaque zero offset: keeps U fragment loads inside the step loop
        // (no hoist+spill) while preserving global address space.
        unsigned uoff = 0;
        asm volatile("" : "+v"(uoff));
        const _Float16* Us = U + uoff;

        // Preload B fragments for this wave's first n-tile (long-latency first)
        v16h bf[8];
#pragma unroll
        for (int kk = 0; kk < 8; ++kk)
            bf[kk] = load_b_packed(Us, GATE >> 4, kk, wid * 4, lane);

        // xW[t] values for this wave's 4 n-tiles (consumed after WMMA chains)
        float xv0[4], xv1[4];
        if (lane < 16) {
#pragma unroll
            for (int nt = 0; nt < 4; ++nt) {
                int col = (wid * 4 + nt) * 16 + lane;
                xv0[nt] = xw[(size_t)t * GATE + col];
                xv1[nt] = xw[(size_t)(LSEQ + t) * GATE + col];
            }
        }

        // A fragments from h (rows 0,1 = batches; rows 2..15 zero)
        v16h af[8];
#pragma unroll
        for (int kk = 0; kk < 8; ++kk) {
            int k0   = kk * 32;
            int m    = lane & 15;
            int kofs = (lane & 16) ? 8 : 0;
            union { uint4 q[2]; v16h v; } uu;
            uu.q[0] = *(const uint4*)&hh16[m][k0 + kofs];
            uu.q[1] = *(const uint4*)&hh16[m][k0 + kofs + 16];
            af[kk] = uu.v;
        }

        // z[:, wid*64 .. wid*64+63] = h @ U + xW[t], double-buffered over tiles
#pragma unroll
        for (int nt = 0; nt < 4; ++nt) {
            int ntg = wid * 4 + nt;
            v16h bn[8];
            if (nt < 3) {
#pragma unroll
                for (int kk = 0; kk < 8; ++kk)
                    bn[kk] = load_b_packed(Us, GATE >> 4, kk, ntg + 1, lane);
            }
            v8f acc = {};
#pragma unroll
            for (int kk = 0; kk < 8; ++kk)
                acc = __builtin_amdgcn_wmma_f32_16x16x32_f16(false, af[kk], false, bf[kk],
                                                             (short)0, acc, false, false);
            if (lane < 16) {   // rows 0,1 of C live in lanes 0..15: acc[0], acc[1]
                int col = ntg * 16 + lane;
                zb[0][col] = acc[0] + xv0[nt];
                zb[1][col] = acc[1] + xv1[nt];
            }
            if (nt < 3) {
#pragma unroll
                for (int kk = 0; kk < 8; ++kk) bf[kk] = bn[kk];
            }
        }
        __syncthreads();

        // gate update: thread -> (batch, hidden unit)
        {
            int b = tid >> 8, k = tid & 255;
            float zi = zb[b][k], zf = zb[b][k + 256], zg = zb[b][k + 512], zo = zb[b][k + 768];
            float cn = sigm(zf) * csf[b][k] + sigm(zi) * tanhf(zg);
            float hn = sigm(zo) * tanhf(cn);
            bool  mb = msk[b][t] != 0;
            float h2 = mb ? hn : hsf[b][k];
            float c2 = mb ? cn : csf[b][k];
            hsf[b][k] = h2; csf[b][k] = c2; hh16[b][k] = (_Float16)h2;
            enc_h[(size_t)(b * LSEQ + t) * ENCD + dir * HID + k] = (_Float16)h2;
        }
        __syncthreads();
    }
}

// r[o] = buv @ Wc[:,o] + bc[o]
__global__ void k_head_r(const float* __restrict__ buv, const float* __restrict__ Wc,
                         const float* __restrict__ bc, float* __restrict__ r, int N) {
    int o = threadIdx.x;
    if (o >= N) return;
    float s = bc[o];
    for (int e = 0; e < 128; ++e) s += buv[e] * Wc[e * N + o];
    r[o] = s;
}

// eh: out[b,i,j] = sigmoid(P[b,j,0] + Q[b,i,0] + r[0]);  P,Q padded to 16 cols
__global__ void k_fin_sig(const float* __restrict__ P, const float* __restrict__ Q,
                          const float* __restrict__ r, float* __restrict__ out) {
    int idx = blockIdx.x * blockDim.x + threadIdx.x;
    if (idx >= 2 * LSEQ * LSEQ) return;
    int j = idx & 255, i = (idx >> 8) & 255, b = idx >> 16;
    float v = P[(size_t)(b * LSEQ + j) * 16] + Q[(size_t)(b * LSEQ + i) * 16] + r[0];
    out[idx] = 1.0f / (1.0f + __expf(-v));
}

// hh/tt: out[b,i,j,:] = softmax_o(P[b,j,o] + Q[b,i,o] + r[o]); padded to 32 cols
__global__ void k_fin_soft(const float* __restrict__ P, const float* __restrict__ Q,
                           const float* __restrict__ r, float* __restrict__ out) {
    int idx = blockIdx.x * blockDim.x + threadIdx.x;
    if (idx >= 2 * LSEQ * LSEQ) return;
    int j = idx & 255, i = (idx >> 8) & 255, b = idx >> 16;
    const float* pr = P + (size_t)(b * LSEQ + j) * 32;
    const float* qr = Q + (size_t)(b * LSEQ + i) * 32;
    float v[24], mx = -1e30f;
#pragma unroll
    for (int o = 0; o < 24; ++o) { v[o] = pr[o] + qr[o] + r[o]; mx = fmaxf(mx, v[o]); }
    float s = 0.0f;
#pragma unroll
    for (int o = 0; o < 24; ++o) { v[o] = __expf(v[o] - mx); s += v[o]; }
    float inv = 1.0f / s;
    float* ob = out + (size_t)idx * 24;
#pragma unroll
    for (int o = 0; o < 24; ++o) ob[o] = v[o] * inv;
}

// ---------------------------------------------------------------------------
// Host launcher
// ---------------------------------------------------------------------------
static inline void* bump(char*& p, size_t bytes) {
    void* r = p;
    p += (bytes + 255) & ~(size_t)255;
    return r;
}

extern "C" void kernel_launch(void* const* d_in, const int* in_sizes, int n_in,
                              void* d_out, int out_size, void* d_ws, size_t ws_size,
                              hipStream_t stream) {
    (void)in_sizes; (void)n_in; (void)out_size; (void)ws_size;
    const int*   inputs = (const int*)d_in[0];
    const int*   words  = (const int*)d_in[1];
    const float* et     = (const float*)d_in[3];
    const float* wt     = (const float*)d_in[4];
    const float* Wf     = (const float*)d_in[5];
    const float* Uf     = (const float*)d_in[6];
    const float* bf     = (const float*)d_in[7];
    const float* Wb     = (const float*)d_in[8];
    const float* Ub     = (const float*)d_in[9];
    const float* bb     = (const float*)d_in[10];
    float* out = (float*)d_out;

    char* wp = (char*)d_ws;
    _Float16* x_h   = (_Float16*)bump(wp, (size_t)512 * 256 * 2);
    _Float16* Wf_p  = (_Float16*)bump(wp, (size_t)256 * GATE * 2);
    _Float16* Wb_p  = (_Float16*)bump(wp, (size_t)256 * GATE * 2);
    _Float16* Uf_p  = (_Float16*)bump(wp, (size_t)HID * GATE * 2);
    _Float16* Ub_p  = (_Float16*)bump(wp, (size_t)HID * GATE * 2);
    float*    xWf   = (float*)bump(wp, (size_t)512 * GATE * 4);
    float*    xWb   = (float*)bump(wp, (size_t)512 * GATE * 4);
    _Float16* enc_h = (_Float16*)bump(wp, (size_t)512 * ENCD * 2);
    _Float16* uW_p  = (_Float16*)bump(wp, (size_t)ENCD * 128 * 2);
    _Float16* vW_p  = (_Float16*)bump(wp, (size_t)ENCD * 128 * 2);
    _Float16* uvT_p = (_Float16*)bump(wp, (size_t)128 * 128 * 2);
    _Float16* uvB_p = (_Float16*)bump(wp, (size_t)128 * 128 * 2);
    _Float16* cls_p = (_Float16*)bump(wp, (size_t)128 * 32 * 2);
    _Float16* u_h   = (_Float16*)bump(wp, (size_t)512 * 128 * 2);
    _Float16* v_h   = (_Float16*)bump(wp, (size_t)512 * 128 * 2);
    _Float16* t_h   = (_Float16*)bump(wp, (size_t)512 * 128 * 2);
    float*    P     = (float*)bump(wp, (size_t)512 * 32 * 4);
    float*    Q     = (float*)bump(wp, (size_t)512 * 32 * 4);
    float*    rbuf  = (float*)bump(wp, 32 * 4);

    auto pack = [&](const float* s, _Float16* d, int K, int N, int Np) {
        k_pack_b<<<(K * Np + 255) / 256, 256, 0, stream>>>(s, d, K, N, Np);
    };
    auto blocks_for = [](int M, int N) { return ((M / 16) * (N / 16) + 7) / 8; };

    // 1) embeddings -> x (f16)
    k_embed<<<(512 * 256 + 255) / 256, 256, 0, stream>>>(inputs, words, et, wt, x_h);

    // 2) LSTM weights -> packed f16 fragments
    pack(Wf, Wf_p, 256, GATE, GATE); pack(Wb, Wb_p, 256, GATE, GATE);
    pack(Uf, Uf_p, HID, GATE, GATE); pack(Ub, Ub_p, HID, GATE, GATE);

    // 3) xW = x @ W + b  (both directions, 512x1024, K=256)
    k_gemm_f16<false, true, true, false>
        <<<blocks_for(512, GATE), 256, 0, stream>>>(x_h, Wf_p, bf, xWf, nullptr,
                                                    512, GATE, 256);
    k_gemm_f16<false, true, true, false>
        <<<blocks_for(512, GATE), 256, 0, stream>>>(x_h, Wb_p, bb, xWb, nullptr,
                                                    512, GATE, 256);

    // 4) recurrence (2 persistent blocks: fwd + bwd; 16 waves each)
    k_lstm<<<2, 512, 0, stream>>>(Uf_p, Ub_p, xWf, xWb, inputs, enc_h);

    // 5) heads: out[b,i,j,o] = P[b,j,o] + Q[b,i,o] + r[o], then sigmoid/softmax
    const size_t out_off[3] = {0, (size_t)2 * 256 * 256, (size_t)2 * 256 * 256 * 25};
    for (int hidx = 0; hidx < 3; ++hidx) {
        int base = 11 + hidx * 8;
        const float* huW  = (const float*)d_in[base + 0];
        const float* hub  = (const float*)d_in[base + 1];
        const float* hvW  = (const float*)d_in[base + 2];
        const float* hvb  = (const float*)d_in[base + 3];
        const float* huvW = (const float*)d_in[base + 4];
        const float* huvb = (const float*)d_in[base + 5];
        const float* hcW  = (const float*)d_in[base + 6];
        const float* hcb  = (const float*)d_in[base + 7];
        int outN = (hidx == 0) ? 1 : 24;
        int Np   = (hidx == 0) ? 16 : 32;

        pack(huW, uW_p, ENCD, 128, 128);
        pack(hvW, vW_p, ENCD, 128, 128);
        pack(huvW,             uvT_p, 128, 128, 128);   // Wuv[:128]
        pack(huvW + 128 * 128, uvB_p, 128, 128, 128);   // Wuv[128:]
        pack(hcW, cls_p, 128, outN, Np);
        k_head_r<<<1, 32, 0, stream>>>(huvb, hcW, hcb, rbuf, outN);

        // u = relu(enc@Wu+bu) -> f16;  t = u@Wuv_top -> f16;  P = t@Wc_pad -> f32
        k_gemm_f16<true, true, false, true>
            <<<blocks_for(512, 128), 256, 0, stream>>>(enc_h, uW_p, hub, nullptr, u_h,
                                                       512, 128, ENCD);
        k_gemm_f16<false, false, false, true>
            <<<blocks_for(512, 128), 256, 0, stream>>>(u_h, uvT_p, nullptr, nullptr, t_h,
                                                       512, 128, 128);
        k_gemm_f16<false, false, true, false>
            <<<blocks_for(512, Np), 256, 0, stream>>>(t_h, cls_p, nullptr, P, nullptr,
                                                      512, Np, 128);
        // v = relu(enc@Wv+bv) -> f16;  t = v@Wuv_bot -> f16;  Q = t@Wc_pad -> f32
        k_gemm_f16<true, true, false, true>
            <<<blocks_for(512, 128), 256, 0, stream>>>(enc_h, vW_p, hvb, nullptr, v_h,
                                                       512, 128, ENCD);
        k_gemm_f16<false, false, false, true>
            <<<blocks_for(512, 128), 256, 0, stream>>>(v_h, uvB_p, nullptr, nullptr, t_h,
                                                       512, 128, 128);
        k_gemm_f16<false, false, true, false>
            <<<blocks_for(512, Np), 256, 0, stream>>>(t_h, cls_p, nullptr, Q, nullptr,
                                                      512, Np, 128);

        if (hidx == 0)
            k_fin_sig<<<(2 * 256 * 256 + 255) / 256, 256, 0, stream>>>(P, Q, rbuf, out);
        else
            k_fin_soft<<<(2 * 256 * 256 + 255) / 256, 256, 0, stream>>>(P, Q, rbuf,
                                                                        out + out_off[hidx]);
    }
}
